// Proposed_Module_LocalGNN_4638564679953
// MI455X (gfx1250) — matrix operand order repaired and checked by reading, not compiled, not run on hardware
//
#include <hip/hip_runtime.h>
#include <math.h>

typedef __attribute__((ext_vector_type(16))) __bf16 v16bf;
typedef __attribute__((ext_vector_type(2)))  __bf16 v2bf;
typedef __attribute__((ext_vector_type(8)))  float  v8f;

#define LRELU(v) ((v) >= 0.f ? (v) : 0.2f * (v))

// ---------------------------------------------------------------------------
// WMMA helpers (CDNA5, wave32).  D = A(16x32 bf16) * B(32x16 bf16) + C(f32)
// ---------------------------------------------------------------------------
__device__ __forceinline__ v8f wmma_bf16(v16bf a, v16bf b, v8f c) {
  return __builtin_amdgcn_wmma_f32_16x16x32_bf16(
      /*neg_a=*/false, a, /*neg_b=*/false, b,
      /*c_mod=*/(short)0, c, /*reuse_a=*/false, /*reuse_b=*/false);
}

// A fragment: rows m = lane&15, K per ISA A-layout:
//   K = (v<4?0:16) + (lane>=16?8:0) + 2*(v&3) + h
// Source: bf16 array, row stride ldk elements (K contiguous).
__device__ __forceinline__ v16bf a_frag_bf16(const __bf16* base, int ldk) {
  int lane = threadIdx.x & 31;
  const __bf16* rp = base + (lane & 15) * ldk;
  int kb = (lane & 16) ? 8 : 0;
  v16bf f;
#pragma unroll
  for (int v = 0; v < 8; ++v) {
    int k = ((v & 4) << 2) + kb + ((v & 3) << 1);
    v2bf p = *(const v2bf*)(rp + k);
    f[2 * v]     = p[0];
    f[2 * v + 1] = p[1];
  }
  return f;
}

// B fragment: cols n = lane&15, K per ISA B-layout:
//   K = (v<4?0:16) + (lane>=16?8:0) + 2*(v&3) + h
// Source: bf16 array laid out [n][k], row stride ldk elements.
__device__ __forceinline__ v16bf b_frag_bf16(const __bf16* base, int ldk) {
  int lane = threadIdx.x & 31;
  const __bf16* cp = base + (lane & 15) * ldk;
  int kb = (lane & 16) ? 8 : 0;
  v16bf f;
#pragma unroll
  for (int v = 0; v < 8; ++v) {
    int k = ((v & 4) << 2) + kb + ((v & 3) << 1);
    v2bf p = *(const v2bf*)(cp + k);
    f[2 * v]     = p[0];
    f[2 * v + 1] = p[1];
  }
  return f;
}

// ---------------------------------------------------------------------------
// prep: repack conv weights [oc][ic][27] f32 -> [tap][oc][ic] bf16,
//       convert adjacency weights to bf16, zero instance-norm stats.
// ---------------------------------------------------------------------------
__global__ void prep_kernel(const float* __restrict__ w_cc1,
                            const float* __restrict__ w_cc2,
                            const float* __restrict__ w_adj1,
                            const float* __restrict__ w_adj2,
                            __bf16* __restrict__ wp1, __bf16* __restrict__ wp2,
                            __bf16* __restrict__ wa1, __bf16* __restrict__ wa2,
                            float* __restrict__ stats) {
  int idx = blockIdx.x * 256 + threadIdx.x;
  if (idx < 55296) {                       // wp1: 27*64*32
    int tap = idx / 2048;
    int oc  = (idx >> 5) & 63;
    int ic  = idx & 31;
    wp1[idx] = (__bf16)w_cc1[(oc * 32 + ic) * 27 + tap];
  } else if (idx < 110592) {               // wp2: 27*32*64
    int i2  = idx - 55296;
    int tap = i2 / 2048;
    int oc  = (i2 >> 6) & 31;
    int ic  = i2 & 63;
    wp2[i2] = (__bf16)w_cc2[(oc * 64 + ic) * 27 + tap];
  } else if (idx < 112640) {               // wa1: 64*32 (already [oc][ic])
    int i2 = idx - 110592;
    wa1[i2] = (__bf16)w_adj1[i2];
  } else if (idx < 118784) {               // wa2: 96*64
    int i2 = idx - 112640;
    wa2[i2] = (__bf16)w_adj2[i2];
  } else if (idx < 118976) {               // stats zero (192 floats)
    stats[idx - 118784] = 0.f;
  }
}

// ---------------------------------------------------------------------------
// 3x3x3 conv, stride1 pad1, implicit GEMM via WMMA.
// Each wave owns one 16-voxel M-tile and ALL OC columns (NACC accumulators),
// so one A fragment per tap feeds NACC WMMAs.  Input slab staged in LDS as
// bf16 (converted once), so A fragments are pure ds_load_b128 of packed bf16.
// ---------------------------------------------------------------------------
template <int IC, int OC, int MT, int MTILES>
__global__ __launch_bounds__(32 * MTILES)
void conv3x3_kernel(const float* __restrict__ in, const __bf16* __restrict__ wp,
                    const float* __restrict__ bias, float* __restrict__ out) {
  constexpr int NTHR = 32 * MTILES;
  constexpr int NACC = OC / 16;
  constexpr int XT = MT + 2;
  __shared__ __bf16 slab[3 * 3 * XT * IC];
  __shared__ float  dtile[OC * (MT + 1)];

  int blk = blockIdx.x;
  int xc = blk % (64 / MT);
  int y  = (blk / (64 / MT)) % 64;
  int z  = blk / ((64 / MT) * 64);
  int x0 = xc * MT;
  int tid = threadIdx.x;

  // stage padded input slab: [z3][y3][x(XT)][ic], ic contiguous, bf16
  for (int idx = tid; idx < 3 * 3 * XT * IC; idx += NTHR) {
    int ic = idx / (3 * 3 * XT);
    int r  = idx % (3 * 3 * XT);
    int zi = r / (3 * XT);
    int r2 = r % (3 * XT);
    int yi = r2 / XT, xi = r2 % XT;
    int gz = z + zi - 1, gy = y + yi - 1, gx = x0 + xi - 1;
    float v = 0.f;
    if ((unsigned)gz < 64u && (unsigned)gy < 64u && (unsigned)gx < 64u)
      v = in[((size_t)ic << 18) + ((size_t)gz << 12) + (gy << 6) + gx];
    slab[((zi * 3 + yi) * XT + xi) * IC + ic] = (__bf16)v;
  }
  __syncthreads();

  int mt = tid >> 5;  // wave id == M-tile id
  v8f acc[NACC];
#pragma unroll
  for (int a = 0; a < NACC; ++a) acc[a] = (v8f){};

  for (int tap = 0; tap < 27; ++tap) {
    int dz = tap / 9, dy = (tap / 3) % 3, dx = tap % 3;
    const __bf16* abase = slab + ((dz * 3 + dy) * XT + mt * 16 + dx) * IC;
    const __bf16* bbase = wp + (size_t)tap * OC * IC;
#pragma unroll
    for (int kc = 0; kc < IC; kc += 32) {
      v16bf a = a_frag_bf16(abase + kc, IC);
#pragma unroll
      for (int na = 0; na < NACC; ++na) {
        v16bf bb = b_frag_bf16(bbase + na * 16 * IC + kc, IC);
        acc[na] = wmma_bf16(a, bb, acc[na]);
      }
    }
  }
  {
    int lane = tid & 31;
    int nb = lane & 15;
    int mb = mt * 16 + ((lane & 16) ? 8 : 0);
#pragma unroll
    for (int na = 0; na < NACC; ++na) {
      int n = na * 16 + nb;
#pragma unroll
      for (int r = 0; r < 8; ++r)
        dtile[n * (MT + 1) + mb + r] = acc[na][r];
    }
  }
  __syncthreads();
  for (int idx = tid; idx < OC * MT; idx += NTHR) {
    int oc = idx / MT, m = idx % MT;
    out[((size_t)oc << 18) + ((size_t)z << 12) + (y << 6) + x0 + m] =
        dtile[oc * (MT + 1) + m] + bias[oc];
  }
}

// ---------------------------------------------------------------------------
// Instance-norm stats (sum, sumsq per channel) + in-place normalize+lrelu
// ---------------------------------------------------------------------------
__global__ __launch_bounds__(256)
void stats_kernel(const float* __restrict__ x, float* __restrict__ sums,
                  float* __restrict__ sqs) {
  int blk = blockIdx.x;
  int ch = blk >> 6, chunk = blk & 63;
  const float* p = x + ((size_t)ch << 18) + chunk * 4096;
  float s = 0.f, q = 0.f;
  for (int i = threadIdx.x; i < 4096; i += 256) {
    float v = p[i];
    s += v;
    q += v * v;
  }
  __shared__ float ss[256], qq[256];
  ss[threadIdx.x] = s;
  qq[threadIdx.x] = q;
  __syncthreads();
  for (int st = 128; st > 0; st >>= 1) {
    if (threadIdx.x < st) {
      ss[threadIdx.x] += ss[threadIdx.x + st];
      qq[threadIdx.x] += qq[threadIdx.x + st];
    }
    __syncthreads();
  }
  if (threadIdx.x == 0) {
    atomicAdd(&sums[ch], ss[0]);
    atomicAdd(&sqs[ch], qq[0]);
  }
}

__global__ __launch_bounds__(256)
void norm_kernel(float* __restrict__ x, const float* __restrict__ sums,
                 const float* __restrict__ sqs) {
  size_t idx = (size_t)blockIdx.x * 256 + threadIdx.x;
  int ch = (int)(idx >> 18);
  const float invN = 1.f / 262144.f;
  float m   = sums[ch] * invN;
  float var = sqs[ch] * invN - m * m;
  float inv = rsqrtf(var + 1e-5f);
  float v = (x[idx] - m) * inv;
  x[idx] = LRELU(v);
}

// ---------------------------------------------------------------------------
// Fused per-window: downsample conv + BN + lrelu, adjacency GNN (WMMA GEMMs),
// softmax, GCN, transposed-conv upsample + BN + lrelu, scatter to output.
// One block (128 threads / 4 waves) per 4^3 window; 4096 windows.
// WMMA A operands (dif, h1) are kept in LDS as bf16.
// ---------------------------------------------------------------------------
__global__ __launch_bounds__(128)
void gnn_kernel(const float* __restrict__ h2n,
                const float* __restrict__ w_down, const float* __restrict__ b_down,
                const float* __restrict__ g_down, const float* __restrict__ be_down,
                const __bf16* __restrict__ wa1, const float* __restrict__ b_adj1,
                const float* __restrict__ g_adj1, const float* __restrict__ be_adj1,
                const __bf16* __restrict__ wa2, const float* __restrict__ b_adj2,
                const float* __restrict__ g_adj2, const float* __restrict__ be_adj2,
                const float* __restrict__ w_adj3, const float* __restrict__ b_adj3,
                const float* __restrict__ gcn_w,
                const float* __restrict__ w_up, const float* __restrict__ b_up,
                const float* __restrict__ g_up, const float* __restrict__ be_up,
                float* __restrict__ out) {
  __shared__ float  win[32 * 64];   // window [ic][voxel]
  __shared__ float  xn[8 * 32];     // node features [n][c]
  __shared__ __bf16 difb[64 * 32];  // |x_i - x_j|  (WMMA A operand)
  __shared__ __bf16 h1b[64 * 64];   // h1           (WMMA A operand)
  __shared__ float  h2s[64 * 96];   // [ij][96]
  __shared__ float  sbuf[64];
  __shared__ float  Pbuf[64];
  __shared__ float  Pxs[8 * 32];
  __shared__ float  gs[8 * 32];

  int b = blockIdx.x;
  int gdi = b & 15, gwi = (b >> 4) & 15, ghi = b >> 8;
  int baseH = ghi * 4, baseW = gwi * 4, baseD = gdi * 4;
  int tid = threadIdx.x;
  const size_t CS = 262144;

  // 1) stage window [ic][hl*16+wl*4+dl]
  for (int idx = tid; idx < 32 * 64; idx += 128) {
    int ic = idx >> 6, v = idx & 63;
    int hl = v >> 4, wl = (v >> 2) & 3, dl = v & 3;
    win[idx] = h2n[(size_t)ic * CS + (size_t)(baseH + hl) * 4096 +
                   (baseW + wl) * 64 + (baseD + dl)];
  }
  __syncthreads();

  // 2) downsample conv k2 s2 + BN + lrelu -> xn[n][oc]
  for (int idx = tid; idx < 256; idx += 128) {
    int n = idx >> 5, oc = idx & 31;
    int p = n >> 2, q = (n >> 1) & 1, r = n & 1;
    float acc = b_down[oc];
    for (int ic = 0; ic < 32; ++ic) {
      const float* wd = w_down + (oc * 32 + ic) * 8;
      const float* wv = win + ic * 64;
#pragma unroll
      for (int t = 0; t < 8; ++t) {
        int i = t >> 2, j = (t >> 1) & 1, k = t & 1;
        acc += wv[((2 * p + i) * 4 + (2 * q + j)) * 4 + (2 * r + k)] * wd[t];
      }
    }
    float val = acc * g_down[oc] + be_down[oc];
    xn[idx] = LRELU(val);
  }
  __syncthreads();

  // 3) dif[i*8+j][c] = |x_i - x_j|  (bf16)
  for (int idx = tid; idx < 2048; idx += 128) {
    int ij = idx >> 5, c = idx & 31;
    int i = ij >> 3, j = ij & 7;
    difb[idx] = (__bf16)fabsf(xn[i * 32 + c] - xn[j * 32 + c]);
  }
  __syncthreads();

  int wave = tid >> 5;
  int lane = tid & 31;

  // 4) h1 = lrelu(bn(dif @ wa1^T + b)) : M=64 N=64 K=32
  for (int t = wave; t < 16; t += 4) {
    int mt = t >> 2, nt = t & 3;
    v8f acc = {};
    v16bf a  = a_frag_bf16(difb + mt * 16 * 32, 32);
    v16bf bb = b_frag_bf16(wa1 + nt * 16 * 32, 32);
    acc = wmma_bf16(a, bb, acc);
    int n  = nt * 16 + (lane & 15);
    int mb = mt * 16 + ((lane & 16) ? 8 : 0);
    float bi = b_adj1[n], ga = g_adj1[n], be = be_adj1[n];
#pragma unroll
    for (int r = 0; r < 8; ++r) {
      float vv = (acc[r] + bi) * ga + be;
      h1b[(mb + r) * 64 + n] = (__bf16)LRELU(vv);
    }
  }
  __syncthreads();

  // 5) h2 = lrelu(bn(h1 @ wa2^T + b)) : M=64 N=96 K=64
  for (int t = wave; t < 24; t += 4) {
    int mt = t / 6, nt = t % 6;
    v8f acc = {};
#pragma unroll
    for (int kc = 0; kc < 64; kc += 32) {
      v16bf a  = a_frag_bf16(h1b + mt * 16 * 64 + kc, 64);
      v16bf bb = b_frag_bf16(wa2 + nt * 16 * 64 + kc, 64);
      acc = wmma_bf16(a, bb, acc);
    }
    int n  = nt * 16 + (lane & 15);
    int mb = mt * 16 + ((lane & 16) ? 8 : 0);
    float bi = b_adj2[n], ga = g_adj2[n], be = be_adj2[n];
#pragma unroll
    for (int r = 0; r < 8; ++r) {
      float vv = (acc[r] + bi) * ga + be;
      h2s[(mb + r) * 96 + n] = LRELU(vv);
    }
  }
  __syncthreads();

  // 6) s[ij] = h2 . w_adj3 + b
  if (tid < 64) {
    float acc = b_adj3[0];
    for (int c = 0; c < 96; ++c) acc += h2s[tid * 96 + c] * w_adj3[c];
    sbuf[tid] = acc;
  }
  __syncthreads();

  // 7) P[i][j] = softmax_j( s[j*8+i] - (i==j)*1e8 )
  if (tid < 8) {
    int i = tid;
    float vals[8];
    float mx = -1e30f;
#pragma unroll
    for (int j = 0; j < 8; ++j) {
      float v = sbuf[j * 8 + i] - (i == j ? 1e8f : 0.f);
      vals[j] = v;
      mx = fmaxf(mx, v);
    }
    float sum = 0.f;
#pragma unroll
    for (int j = 0; j < 8; ++j) {
      vals[j] = expf(vals[j] - mx);
      sum += vals[j];
    }
    float inv = 1.f / sum;
#pragma unroll
    for (int j = 0; j < 8; ++j) Pbuf[i * 8 + j] = vals[j] * inv;
  }
  __syncthreads();

  // 8) Px[i][c] = sum_j P[i][j] x[j][c]
  for (int idx = tid; idx < 256; idx += 128) {
    int i = idx >> 5, c = idx & 31;
    float acc = 0.f;
#pragma unroll
    for (int j = 0; j < 8; ++j) acc += Pbuf[i * 8 + j] * xn[j * 32 + c];
    Pxs[idx] = acc;
  }
  __syncthreads();

  // 9) g = lrelu([x, Px] @ gcn_w)   (gcn_w [64][32])
  for (int idx = tid; idx < 256; idx += 128) {
    int i = idx >> 5, oc = idx & 31;
    float acc = 0.f;
    for (int f = 0; f < 32; ++f) acc += xn[i * 32 + f] * gcn_w[f * 32 + oc];
    for (int f = 0; f < 32; ++f) acc += Pxs[i * 32 + f] * gcn_w[(32 + f) * 32 + oc];
    gs[idx] = LRELU(acc);
  }
  __syncthreads();

  // 10) transposed conv k2 s2 + BN + lrelu, scatter to global out
  for (int idx = tid; idx < 2048; idx += 128) {
    int oc = idx >> 6, v = idx & 63;
    int hl = v >> 4, wl = (v >> 2) & 3, dl = v & 3;
    int n = ((hl >> 1) * 2 + (wl >> 1)) * 2 + (dl >> 1);
    int i = hl & 1, j = wl & 1, k = dl & 1;
    float acc = b_up[oc];
    for (int ic = 0; ic < 32; ++ic)
      acc += gs[n * 32 + ic] * w_up[(((ic * 32 + oc) * 2 + i) * 2 + j) * 2 + k];
    float val = acc * g_up[oc] + be_up[oc];
    val = LRELU(val);
    out[(size_t)oc * CS + (size_t)(baseH + hl) * 4096 + (baseW + wl) * 64 +
        (baseD + dl)] = val;
  }
}

// ---------------------------------------------------------------------------
extern "C" void kernel_launch(void* const* d_in, const int* in_sizes, int n_in,
                              void* d_out, int out_size, void* d_ws, size_t ws_size,
                              hipStream_t stream) {
  (void)in_sizes; (void)n_in; (void)out_size; (void)ws_size;
  const float* x_concat = (const float*)d_in[0];
  const float* w_cc1  = (const float*)d_in[1];
  const float* b_cc1  = (const float*)d_in[2];
  const float* w_cc2  = (const float*)d_in[3];
  const float* b_cc2  = (const float*)d_in[4];
  const float* w_down = (const float*)d_in[5];
  const float* b_down = (const float*)d_in[6];
  const float* g_down = (const float*)d_in[7];
  const float* be_down = (const float*)d_in[8];
  const float* w_adj1 = (const float*)d_in[9];
  const float* b_adj1 = (const float*)d_in[10];
  const float* g_adj1 = (const float*)d_in[11];
  const float* be_adj1 = (const float*)d_in[12];
  const float* w_adj2 = (const float*)d_in[13];
  const float* b_adj2 = (const float*)d_in[14];
  const float* g_adj2 = (const float*)d_in[15];
  const float* be_adj2 = (const float*)d_in[16];
  const float* w_adj3 = (const float*)d_in[17];
  const float* b_adj3 = (const float*)d_in[18];
  const float* gcn_w  = (const float*)d_in[19];
  const float* w_up   = (const float*)d_in[20];
  const float* b_up   = (const float*)d_in[21];
  const float* g_up   = (const float*)d_in[22];
  const float* be_up  = (const float*)d_in[23];
  float* out = (float*)d_out;

  // workspace layout
  float* buf1  = (float*)d_ws;              // conv1 out, 64*262144 f32
  float* buf2  = buf1 + 16777216;           // conv2 out, 32*262144 f32
  float* stats = buf2 + 8388608;            // 192 f32: s1[64] q1[64] s2[32] q2[32]
  __bf16* wp1 = (__bf16*)(stats + 192);     // 27*64*32
  __bf16* wp2 = wp1 + 55296;                // 27*32*64
  __bf16* wa1 = wp2 + 55296;                // 64*32
  __bf16* wa2 = wa1 + 2048;                 // 96*64

  prep_kernel<<<465, 256, 0, stream>>>(w_cc1, w_cc2, w_adj1, w_adj2,
                                       wp1, wp2, wa1, wa2, stats);

  // conv1: 32 -> 64 channels.  MT=64, 4 waves, 4 accumulators/wave.
  conv3x3_kernel<32, 64, 64, 4><<<4096, 128, 0, stream>>>(x_concat, wp1, b_cc1, buf1);
  stats_kernel<<<64 * 64, 256, 0, stream>>>(buf1, stats, stats + 64);
  norm_kernel<<<65536, 256, 0, stream>>>(buf1, stats, stats + 64);

  // conv2: 64 -> 32 channels.  MT=32, 2 waves, 2 accumulators/wave.
  conv3x3_kernel<64, 32, 32, 2><<<8192, 64, 0, stream>>>(buf1, wp2, b_cc2, buf2);
  stats_kernel<<<32 * 64, 256, 0, stream>>>(buf2, stats + 128, stats + 160);
  norm_kernel<<<32768, 256, 0, stream>>>(buf2, stats + 128, stats + 160);

  // per-window GNN + upsample -> out
  gnn_kernel<<<4096, 128, 0, stream>>>(buf2,
      w_down, b_down, g_down, be_down,
      wa1, b_adj1, g_adj1, be_adj1,
      wa2, b_adj2, g_adj2, be_adj2,
      w_adj3, b_adj3, gcn_w,
      w_up, b_up, g_up, be_up, out);
}
